// PointnetSAModuleMSGCG_81037442941236
// MI455X (gfx1250) — matrix operand order repaired and checked by reading, not compiled, not run on hardware
//
#include <hip/hip_runtime.h>
#include <hip/hip_bf16.h>
#include <math.h>

// ---------------------------------------------------------------------------
// PointNet++ MSG SA module for MI455X (gfx1250, wave32, WMMA).
// All shared-MLP 1x1 convs run as bf16 WMMA GEMMs (v_wmma_f32_16x16x32_bf16)
// with 2x4 register blocking (each wave owns a 32x64 output block = 8 wmma
// per 32-wide K step from 6 fragment loads). A/B/D operands are kept
// *fragment-packed in memory* so fragments load as contiguous b128 vectors:
//   XT[b][kt][nt][lane][e]  : B fragments (32x16 bf16 tiles, K x N)
//   WT[ot][kt][lane][e]     : A fragments (16x32 bf16 tiles, M x K)
//   ZT[b][ot][nt][lane][v]  : D fragments (16x16 f32 -> bf16, 16B/lane store)
// BN is training-mode (global stats) -> two-phase per layer with a
// deterministic shuffle-tree + fixed-order reduction (replay-stable).
// Workspace requirement: ~290 MB.
// ---------------------------------------------------------------------------

typedef __attribute__((ext_vector_type(16))) __bf16 v16bf;
typedef __attribute__((ext_vector_type(8)))  float  v8f;
typedef unsigned short u16;
typedef unsigned int   u32;

#define B_   4
#define N_   8192
#define S_   1024
#define CIN_ 128
#define EPS_ 1e-5f

__device__ __forceinline__ float bf2f(u16 h) {
    u32 u = ((u32)h) << 16;
    return __builtin_bit_cast(float, u);
}
__device__ __forceinline__ u16 f2bf(float f) {
    u32 u = __builtin_bit_cast(u32, f);
    return (u16)((u + 0x7FFFu + ((u >> 16) & 1u)) >> 16);   // RNE
}

// XT element offset (u16 units) for logical (b, channel c, point p)
__device__ __forceinline__ size_t xt_off(int b, int c, int p, int KT, int NT) {
    int kt = c >> 5, r = c & 31, half = r >> 4, e = r & 15;
    int lane = (half << 4) | (p & 15), nt = p >> 4;
    return ((((size_t)b * KT + kt) * NT + nt) * 32 + lane) * 16 + e;
}
// ZT element offset (u16 units) for logical (b, channel c, point p)
__device__ __forceinline__ size_t zt_off(int b, int c, int p, int OT, int NT) {
    int ot = c >> 4, m = c & 15, v = m & 7, half = m >> 3;
    int lane = (half << 4) | (p & 15), nt = p >> 4;
    return ((((size_t)b * OT + ot) * NT + nt) * 32 + lane) * 8 + v;
}

// -------------------------------------------------------------------- ctr copy
__global__ void ctr_copy_kernel(const float* __restrict__ ctr, float* __restrict__ out) {
    int t = blockIdx.x * blockDim.x + threadIdx.x;
    if (t < B_ * S_ * 3) out[t] = ctr[t];
}

// --------------------------------------------- weight pack into A-fragment form
__global__ void wpack_kernel(const float* __restrict__ W, u16* __restrict__ WT,
                             int Oreal, int Creal, int OT, int KT) {
    int t = blockIdx.x * blockDim.x + threadIdx.x;
    if (t >= OT * KT * 512) return;
    int e = t & 15, lane = (t >> 4) & 31;
    int kt = (t >> 9) % KT, ot = (t >> 9) / KT;
    int v = e >> 1, pb = e & 1;
    // A (16x32 bf16, MxK): lanes0-15 K0-15, lanes16-31 K16-31 (k = 16*half + e)
    int ka = ((lane >> 4) << 4) + e; (void)v; (void)pb;
    int o = ot * 16 + (lane & 15);
    int c = kt * 32 + ka;
    float val = (o < Oreal && c < Creal) ? W[(size_t)o * Creal + c] : 0.0f;
    WT[t] = f2bf(val);
}

// ------------------------------------------------------------------ ball query
// One wave32 per center; ballot-based ordered compaction reproduces pointnet2
// ball_query semantics (first nsample ascending in-radius idx, pad w/ first).
__global__ void ballquery_kernel(const float* __restrict__ xyz,
                                 const float* __restrict__ ctr,
                                 int* __restrict__ idx0, int* __restrict__ idx1,
                                 int* __restrict__ idx2) {
    int tid = threadIdx.x, lane = tid & 31, wid = tid >> 5;
    int w = blockIdx.x * 8 + wid;            // 0 .. B*S-1
    if (w >= B_ * S_) return;
    int b = w / S_;
    const float* c = ctr + (size_t)w * 3;
    float cx = c[0], cy = c[1], cz = c[2];

    const float r2[3] = {0.01f, 0.04f, 0.16f};
    const int   Ks[3] = {16, 32, 64};
    int*        op[3] = {idx0 + (size_t)w * 16, idx1 + (size_t)w * 32, idx2 + (size_t)w * 64};
    int cnt[3] = {0, 0, 0}, frst[3] = {0, 0, 0}, hasf[3] = {0, 0, 0};

    for (int p0 = 0; p0 < N_; p0 += 32) {
        int p = p0 + lane;
        const float* q = xyz + ((size_t)b * N_ + p) * 3;
        float dx = q[0] - cx, dy = q[1] - cy, dz = q[2] - cz;
        float d2 = dx * dx + dy * dy + dz * dz;
        for (int i = 0; i < 3; ++i) {
            if (cnt[i] >= Ks[i]) continue;           // uniform across wave
            bool valid = d2 < r2[i];
            unsigned m = (unsigned)__ballot(valid);
            if (m) {
                if (!hasf[i]) { hasf[i] = 1; frst[i] = p0 + (__ffs((int)m) - 1); }
                int pos = cnt[i] + __popc(m & ((1u << lane) - 1u));
                if (valid && pos < Ks[i]) op[i][pos] = p;
                cnt[i] += __popc(m);
            }
        }
        if (cnt[0] >= 16 && cnt[1] >= 32 && cnt[2] >= 64) break;
    }
    for (int i = 0; i < 3; ++i) {
        int fill  = hasf[i] ? frst[i] : 0;
        int start = cnt[i] < Ks[i] ? cnt[i] : Ks[i];
        for (int j = start + lane; j < Ks[i]; j += 32) op[i][j] = fill;
    }
}

// --------------------------------------------------- gather into B-fragments
// Builds XT for Cpad=160: ch 0..2 rel-xyz, 3..130 features, 131..159 zero.
__global__ void gather_pack_kernel(const float* __restrict__ xyz,
                                   const float* __restrict__ feat,
                                   const float* __restrict__ ctr,
                                   const int* __restrict__ idx,
                                   u16* __restrict__ XT, int K, int NT, int total) {
    int t = blockIdx.x * blockDim.x + threadIdx.x;
    if (t >= total) return;
    int k = t % K, s = (t / K) % S_, b = t / (K * S_);
    int p = s * K + k;
    int id = idx[t];
    const float* q = xyz + ((size_t)b * N_ + id) * 3;
    const float* c = ctr + ((size_t)b * S_ + s) * 3;
    XT[xt_off(b, 0, p, 5, NT)] = f2bf(q[0] - c[0]);
    XT[xt_off(b, 1, p, 5, NT)] = f2bf(q[1] - c[1]);
    XT[xt_off(b, 2, p, 5, NT)] = f2bf(q[2] - c[2]);
    const float* fb = feat + (size_t)b * CIN_ * N_ + id;
    __builtin_prefetch(fb, 0, 3);                         // global_prefetch_b8
#pragma unroll 4
    for (int j = 0; j < CIN_; ++j)
        XT[xt_off(b, 3 + j, p, 5, NT)] = f2bf(fb[(size_t)j * N_]);
    for (int j = 131; j < 160; ++j) XT[xt_off(b, j, p, 5, NT)] = 0;
}

// ------------------------------------------------------ WMMA GEMM (packed IO)
// ZT = WT x XT (+bias). Block = 8 waves; each wave owns a 32(M) x 64(N) block
// (2 ot-tiles x 4 nt-tiles): per 32-K step, 6 fragment loads feed 8 wmmas.
__global__ void gemm_wmma_bf16(const v16bf* __restrict__ WT,
                               const v16bf* __restrict__ XT,
                               uint4* __restrict__ ZT,
                               float* __restrict__ part,
                               const float* __restrict__ bias,
                               int KT, int NT, int Oreal) {
    __shared__ float wsum[8][2][16];
    __shared__ float wsq[8][2][16];
    int tid = threadIdx.x, lane = tid & 31, wid = tid >> 5;
    int half = lane >> 4;
    int b   = blockIdx.z;
    int ot0 = blockIdx.y * 2;                 // M pair
    int OT  = gridDim.y * 2;
    int ntb = (blockIdx.x * 8 + wid) * 4;     // N quad

    const v16bf* Ap = WT + ((size_t)ot0 * KT) * 32 + lane;           // ot1: +KT*32
    const v16bf* Bp = XT + (((size_t)b * KT) * NT + ntb) * 32 + lane;
    size_t bstride = (size_t)NT * 32;

    v8f acc[2][4];
#pragma unroll
    for (int a = 0; a < 2; ++a)
#pragma unroll
        for (int i = 0; i < 4; ++i)
            acc[a][i] = (v8f){0.f, 0.f, 0.f, 0.f, 0.f, 0.f, 0.f, 0.f};

    for (int kt = 0; kt < KT; ++kt) {
        v16bf a0 = Ap[(size_t)kt * 32];
        v16bf a1 = Ap[((size_t)KT + kt) * 32];
        v16bf bf[4];
#pragma unroll
        for (int i = 0; i < 4; ++i) bf[i] = Bp[(size_t)kt * bstride + i * 32];
        __builtin_prefetch(Bp + (size_t)(kt + 1) * bstride, 0, 3);
#pragma unroll
        for (int i = 0; i < 4; ++i) {
            acc[0][i] = __builtin_amdgcn_wmma_f32_16x16x32_bf16(false, a0, false, bf[i],
                                                                (short)0, acc[0][i], false, false);
            acc[1][i] = __builtin_amdgcn_wmma_f32_16x16x32_bf16(false, a1, false, bf[i],
                                                                (short)0, acc[1][i], false, false);
        }
    }

    // epilogue: bias, packed b128 stores, deterministic stat partials
    float zs[2][8], zq[2][8];
#pragma unroll
    for (int a = 0; a < 2; ++a)
#pragma unroll
        for (int v = 0; v < 8; ++v) { zs[a][v] = 0.f; zq[a][v] = 0.f; }

#pragma unroll
    for (int a = 0; a < 2; ++a) {
        float bv[8];
#pragma unroll
        for (int v = 0; v < 8; ++v) {
            int ch = (ot0 + a) * 16 + v + half * 8;
            bv[v] = (bias != nullptr && ch < Oreal) ? bias[ch] : 0.f;
        }
#pragma unroll
        for (int i = 0; i < 4; ++i) {
            float z[8];
#pragma unroll
            for (int v = 0; v < 8; ++v) {
                z[v] = acc[a][i][v] + bv[v];
                zs[a][v] += z[v];
                zq[a][v] += z[v] * z[v];
            }
            uint4 pk;
            pk.x = (u32)f2bf(z[0]) | ((u32)f2bf(z[1]) << 16);
            pk.y = (u32)f2bf(z[2]) | ((u32)f2bf(z[3]) << 16);
            pk.z = (u32)f2bf(z[4]) | ((u32)f2bf(z[5]) << 16);
            pk.w = (u32)f2bf(z[6]) | ((u32)f2bf(z[7]) << 16);
            ZT[(((size_t)b * OT + ot0 + a) * NT + ntb + i) * 32 + lane] = pk;
        }
    }
    // in-half shuffle tree over the 16 N-lanes (deterministic, replay-stable)
#pragma unroll
    for (int a = 0; a < 2; ++a)
#pragma unroll
        for (int v = 0; v < 8; ++v) {
            float sv = zs[a][v], sq = zq[a][v];
            for (int off = 8; off; off >>= 1) {
                sv += __shfl_xor(sv, off, 16);
                sq += __shfl_xor(sq, off, 16);
            }
            if ((lane & 15) == 0) {
                wsum[wid][a][v + half * 8] = sv;
                wsq[wid][a][v + half * 8]  = sq;
            }
        }
    __syncthreads();
    if (tid < 32) {
        int a = tid >> 4, m = tid & 15;
        float s = 0.f, q = 0.f;
        for (int wv = 0; wv < 8; ++wv) { s += wsum[wv][a][m]; q += wsq[wv][a][m]; }
        size_t blin = ((size_t)blockIdx.z * gridDim.y + blockIdx.y) * gridDim.x + blockIdx.x;
        part[blin * 64 + a * 32 + m]      = s;
        part[blin * 64 + a * 32 + 16 + m] = q;
    }
}

// --------------------------------------------------- deterministic stat reduce
__global__ void reduce_stats_kernel(const float* __restrict__ part,
                                    float* __restrict__ stats, int gx, int OT) {
    int c = blockIdx.x * blockDim.x + threadIdx.x;
    if (c >= OT * 16) return;
    int ot = c >> 4, m = c & 15;
    int by = ot >> 1, a = ot & 1, gy = OT >> 1;
    float s = 0.f, q = 0.f;
    for (int b = 0; b < B_; ++b)
        for (int x = 0; x < gx; ++x) {
            size_t lin = ((size_t)b * gy + by) * gx + x;
            s += part[lin * 64 + a * 32 + m];
            q += part[lin * 64 + a * 32 + 16 + m];
        }
    stats[c]       = s;
    stats[512 + c] = q;
}

// ------------------------------------- BN + ReLU: ZT -> XT (next layer input)
// Thread index is the *output* XT offset -> fully coalesced writes.
__global__ void bnrelu_pack_kernel(const u16* __restrict__ ZT, u16* __restrict__ XT,
                                   const float* __restrict__ stats,
                                   const float* __restrict__ gamma,
                                   const float* __restrict__ beta,
                                   int Creal, int OTz, int KTn, int NT,
                                   float invCnt, int total) {
    int t = blockIdx.x * blockDim.x + threadIdx.x;
    if (t >= total) return;
    int e = t & 15, lane = (t >> 4) & 31;
    int u = t >> 9;
    int nt = u % NT; u /= NT;
    int kt = u % KTn; int b = u / KTn;
    int c = kt * 32 + ((lane >> 4) << 4) + e;
    int p = nt * 16 + (lane & 15);
    float y = 0.f;
    if (c < Creal) {
        float zz   = bf2f(ZT[zt_off(b, c, p, OTz, NT)]);
        float mean = stats[c] * invCnt;
        float var  = stats[512 + c] * invCnt - mean * mean;
        float a    = gamma[c] * rsqrtf(var + EPS_);
        y = fmaxf(a * (zz - mean) + beta[c], 0.f);
    }
    XT[t] = f2bf(y);
}

// --------------------------- BN + ReLU + max over K -> featc (B-fragment form)
__global__ void bnrelu_maxpool_kernel(const u16* __restrict__ ZT,
                                      u16* __restrict__ Feat,
                                      const float* __restrict__ stats,
                                      const float* __restrict__ gamma,
                                      const float* __restrict__ beta,
                                      int K, int NT, float invCnt, int coff) {
    int t = blockIdx.x * blockDim.x + threadIdx.x;
    if (t >= B_ * 256 * S_) return;
    int s = t % S_, o = (t / S_) % 256, b = t / (S_ * 256);
    float mean = stats[o] * invCnt;
    float var  = stats[512 + o] * invCnt - mean * mean;
    float a    = gamma[o] * rsqrtf(var + EPS_);
    float sh   = beta[o] - a * mean;
    float mx = 0.f;                                   // relu output >= 0
    for (int k = 0; k < K; ++k) {
        float zz = bf2f(ZT[zt_off(b, o, s * K + k, 16, NT)]);
        mx = fmaxf(mx, fmaxf(a * zz + sh, 0.f));
    }
    // featc is XT-packed for the final GEMM: Cpad=768 (KT=24), P=1024 (NT=64)
    Feat[xt_off(b, coff + o, s, 24, 64)] = f2bf(mx);
}

// ------------------------------------------------------- final BN+ReLU -> f32
__global__ void final_out_kernel(const u16* __restrict__ ZT, float* __restrict__ out,
                                 const float* __restrict__ stats,
                                 const float* __restrict__ gamma,
                                 const float* __restrict__ beta, float invCnt) {
    int t = blockIdx.x * blockDim.x + threadIdx.x;
    if (t >= B_ * 512 * S_) return;
    int s = t % S_, o = (t / S_) % 512, b = t / (S_ * 512);
    float zz   = bf2f(ZT[zt_off(b, o, s, 32, 64)]);
    float mean = stats[o] * invCnt;
    float var  = stats[512 + o] * invCnt - mean * mean;
    float a    = gamma[o] * rsqrtf(var + EPS_);
    out[((size_t)b * 512 + o) * S_ + s] = fmaxf(a * (zz - mean) + beta[o], 0.f);
}

// ===========================================================================
extern "C" void kernel_launch(void* const* d_in, const int* in_sizes, int n_in,
                              void* d_out, int out_size, void* d_ws, size_t ws_size,
                              hipStream_t stream) {
    const float* xyz  = (const float*)d_in[0];
    const float* feat = (const float*)d_in[1];
    const float* ctr  = (const float*)d_in[2];
    // mlp_params flattened: layer l = s*3+j -> d_in[3+l*3+{0:W,1:gamma,2:beta}]
    const float* conv1_w = (const float*)d_in[30];
    const float* conv1_b = (const float*)d_in[31];
    const float* bn1_g   = (const float*)d_in[32];
    const float* bn1_b   = (const float*)d_in[33];
    float* out = (float*)d_out;

    static const int Ks[3] = {16, 32, 64};
    static const int Ps[3] = {S_ * 16, S_ * 32, S_ * 64};
    // Per-layer padded/real dims (layers 0..8 = scale MLPs, layer 9 = final conv)
    // Opad is padded to a multiple of 32 so the GEMM's 2-wide M blocking works.
    static const int wOr[10] = {128, 128, 256, 128, 196, 256, 128, 256, 256, 512};
    static const int wCr[10] = {131, 128, 128, 131, 128, 196, 131, 128, 256, 768};
    static const int wOp[10] = {128, 128, 256, 128, 224, 256, 128, 256, 256, 512};
    static const int wCp[10] = {160, 128, 128, 160, 128, 224, 160, 128, 256, 768};

    // ---------------- workspace layout ----------------
    char* wsb = (char*)d_ws;
    size_t off = 0;
    auto alloc = [&](size_t bytes) -> void* {
        void* p = wsb + off;
        off = (off + bytes + 255) & ~(size_t)255;
        return p;
    };
    int* idxb[3];
    for (int i = 0; i < 3; ++i) idxb[i] = (int*)alloc((size_t)B_ * S_ * Ks[i] * 4);
    u16* wts[10];
    for (int l = 0; l < 10; ++l) wts[l] = (u16*)alloc((size_t)wOp[l] * wCp[l] * 2);
    float* stats = (float*)alloc(10 * 1024 * sizeof(float));   // [layer][sum512|sq512]
    float* part  = (float*)alloc((size_t)512 * 16 * B_ * 64 * sizeof(float));
    size_t bufElems = (size_t)B_ * 256 * Ps[2];                // 67.1M elems
    u16* bufA  = (u16*)alloc(bufElems * 2);                    // 134 MB
    u16* bufB  = (u16*)alloc(bufElems * 2);                    // 134 MB
    u16* featc = (u16*)alloc((size_t)B_ * 768 * S_ * 2);

    // ---------------- setup ----------------
    ctr_copy_kernel<<<(B_ * S_ * 3 + 255) / 256, 256, 0, stream>>>(ctr, out);
    for (int l = 0; l < 10; ++l) {
        const float* W = (l < 9) ? (const float*)d_in[3 + l * 3] : conv1_w;
        int tot = (wOp[l] / 16) * (wCp[l] / 32) * 512;
        wpack_kernel<<<(tot + 255) / 256, 256, 0, stream>>>(W, wts[l], wOr[l], wCr[l],
                                                            wOp[l] / 16, wCp[l] / 32);
    }
    ballquery_kernel<<<(B_ * S_) / 8, 256, 0, stream>>>(xyz, ctr, idxb[0], idxb[1], idxb[2]);

    // ---------------- per-scale shared MLPs ----------------
    for (int s = 0; s < 3; ++s) {
        int K = Ks[s], P = Ps[s], NT = P / 16;
        float invCnt = 1.0f / (float)(B_ * P);
        int totG = B_ * S_ * K;
        gather_pack_kernel<<<(totG + 255) / 256, 256, 0, stream>>>(xyz, feat, ctr, idxb[s],
                                                                   bufA, K, NT, totG);
        for (int j = 0; j < 3; ++j) {
            int l = s * 3 + j;
            int KT = wCp[l] / 32, OT = wOp[l] / 16;
            int gx = NT / 32;
            float* st = stats + l * 1024;
            // bufA holds this layer's XT input; bufB receives ZT
            gemm_wmma_bf16<<<dim3(gx, OT / 2, B_), 256, 0, stream>>>(
                (const v16bf*)wts[l], (const v16bf*)bufA, (uint4*)bufB,
                part, nullptr, KT, NT, wOr[l]);
            reduce_stats_kernel<<<(wOp[l] + 255) / 256, 256, 0, stream>>>(part, st, gx, OT);
            const float* gm = (const float*)d_in[3 + l * 3 + 1];
            const float* bt = (const float*)d_in[3 + l * 3 + 2];
            if (j < 2) {
                int KTn = wCp[l + 1] / 32;             // next layer's padded C / 32
                int tot = B_ * KTn * NT * 512;
                bnrelu_pack_kernel<<<(tot + 255) / 256, 256, 0, stream>>>(
                    bufB, bufA, st, gm, bt, wOr[l], OT, KTn, NT, invCnt, tot);
            } else {
                int tot = B_ * 256 * S_;
                bnrelu_maxpool_kernel<<<(tot + 255) / 256, 256, 0, stream>>>(
                    bufB, featc, st, gm, bt, K, NT, invCnt, 256 * s);
            }
        }
    }

    // ---------------- final 768 -> 512 conv + BN + ReLU ----------------
    {
        int NT = S_ / 16, KT = 768 / 32, OT = 512 / 16, gx = NT / 32;
        float* st = stats + 9 * 1024;
        gemm_wmma_bf16<<<dim3(gx, OT / 2, B_), 256, 0, stream>>>(
            (const v16bf*)wts[9], (const v16bf*)featc, (uint4*)bufA,
            part, conv1_b, KT, NT, 512);
        reduce_stats_kernel<<<(512 + 255) / 256, 256, 0, stream>>>(part, st, gx, OT);
        int tot = B_ * 512 * S_;
        final_out_kernel<<<(tot + 255) / 256, 256, 0, stream>>>(
            bufA, out + (size_t)B_ * S_ * 3, st, bn1_g, bn1_b, 1.0f / (float)(B_ * S_));
    }
    (void)in_sizes; (void)n_in; (void)out_size; (void)ws_size;
}
